// SparseMMoE_78434692759667
// MI455X (gfx1250) — matrix operand order repaired and checked by reading, not compiled, not run on hardware
//
#include <hip/hip_runtime.h>
#include <hip/hip_fp16.h>
#include <math.h>

// Problem constants (match reference)
#define BB 65536
#define DD 512
#define OO 64
#define EE 8
#define TT 2
#define H1V 64
#define H2V 32
#define WPB 2                      // waves per block in the MLP kernel
#define NBLK (BB / (WPB * 16))     // MLP blocks (each wave = 16 tokens)

typedef __attribute__((ext_vector_type(16))) _Float16 v16h;
typedef __attribute__((ext_vector_type(8)))  _Float16 v8h;
typedef __attribute__((ext_vector_type(8)))  float    v8f;
typedef __attribute__((ext_vector_type(4)))  float    v4f;
typedef __attribute__((ext_vector_type(2)))  float    v2f;

// ---------------------------------------------------------------------------
// Kernel 1: convert weights. Expert weights f32 -> f16 transposed to [E][N][K]
// (B-fragment = one contiguous 32B per-lane load); gating weights f32
// transposed to [16 cols=(t*8+e)][D] (kept f32 for exact top-k).
// ---------------------------------------------------------------------------
__global__ __launch_bounds__(256) void convert_weights(
    const float* __restrict__ W1, const float* __restrict__ W2,
    const float* __restrict__ W3, const float* __restrict__ wg,
    _Float16* __restrict__ W1t, _Float16* __restrict__ W2t,
    _Float16* __restrict__ W3t, float* __restrict__ wgt)
{
  int idx = blockIdx.x * 256 + threadIdx.x;
  if (idx < EE * DD * H1V) {              // W1 [E][D][H1] -> W1t [E][H1][D]
    int e = idx / (DD * H1V), r = idx % (DD * H1V);
    int d = r / H1V, h = r % H1V;
    W1t[((size_t)e * H1V + h) * DD + d] = (_Float16)W1[idx];
  }
  if (idx < EE * H1V * H2V) {             // W2 [E][H1][H2] -> W2t [E][H2][H1]
    int e = idx / (H1V * H2V), r = idx % (H1V * H2V);
    int h = r / H2V, o = r % H2V;
    W2t[((size_t)e * H2V + o) * H1V + h] = (_Float16)W2[idx];
  }
  if (idx < EE * H2V * OO) {              // W3 [E][H2][O] -> W3t [E][O][H2]
    int e = idx / (H2V * OO), r = idx % (H2V * OO);
    int h = r / OO, o = r % OO;
    W3t[((size_t)e * OO + o) * H2V + h] = (_Float16)W3[idx];
  }
  if (idx < TT * DD * EE) {               // wg [T][D][E] -> wgt [T*8+e][D] f32
    int t = idx / (DD * EE), r = idx % (DD * EE);
    int d = r / EE, e = r % EE;
    wgt[((size_t)t * EE + e) * DD + d] = wg[idx];
  }
}

// ---- WMMA fragment helpers (f16 path) ----
__device__ __forceinline__ v16h loadA16(const _Float16* base, int ln, int half,
                                        int stride, int c) {
  // A 16x32 f16 layout: lane(&15)=row M; elems 0..15 = K c*32+half*16..+15.
  const _Float16* ap = base + ln * stride + c * 32 + half * 16;
  v8h a0 = *(const v8h*)ap;
  v8h a1 = *(const v8h*)(ap + 8);
  v16h a;
#pragma unroll
  for (int i = 0; i < 8; ++i) { a[i] = a0[i]; a[8 + i] = a1[i]; }
  return a;
}
__device__ __forceinline__ v16h loadB16(const _Float16* w, int stride, int nt,
                                        int ln, int half, int c) {
  // B 32x16 f16 layout: lane(&15)=col N; 16 contiguous K selected by half.
  return *(const v16h*)(w + (size_t)(nt * 16 + ln) * stride + c * 32 + half * 16);
}

// ---------------------------------------------------------------------------
// Kernel 2: fully fused MoE: f32-WMMA gating (exact top-k) + top-2 softmax +
// 8 expert MLPs (f16 WMMA, f32 accum) + gated combine, one pass over x.
// Wave = 16 tokens. x staged once to LDS as f16; layer-1 B loads are
// software-pipelined with sched_barrier; gates live in registers via shuffles.
// importance/load -> LDS reduction -> per-block partials in workspace.
// ---------------------------------------------------------------------------
__global__ __launch_bounds__(WPB * 32) void moe_fused_kernel(
    const float* __restrict__ x,  const float* __restrict__ wgt,
    const float* __restrict__ bg,
    const _Float16* __restrict__ W1t, const float* __restrict__ b1,
    const _Float16* __restrict__ W2t, const float* __restrict__ b2,
    const _Float16* __restrict__ W3t, const float* __restrict__ b3,
    float* __restrict__ y, float* __restrict__ partials)
{
  __shared__ alignas(16) _Float16 xs[WPB][16 * DD];             // 16 tok x 512
  __shared__ alignas(16) _Float16 hs[WPB][16 * H1V + 16 * H2V]; // h1 + h2
  __shared__ float red[32];   // [0..15] importance(t*8+e), [16..31] load

  const int wave = threadIdx.x >> 5;
  const int lane = threadIdx.x & 31;
  const int half = lane >> 4;
  const int ln   = lane & 15;
  const int tok0 = (blockIdx.x * WPB + wave) * 16;

  if (threadIdx.x < 32) red[threadIdx.x] = 0.f;
  __syncthreads();

  _Float16* xw  = &xs[wave][0];
  _Float16* h1s = &hs[wave][0];
  _Float16* h2s = &hs[wave][16 * H1V];

  const v8f zero8 = {0.f, 0.f, 0.f, 0.f, 0.f, 0.f, 0.f, 0.f};

  // --- stage x tile (16 tokens x 512) into LDS as f16 (coalesced rows) ---
#pragma unroll
  for (int r = 0; r < 16; ++r) {
    const float* p = x + (size_t)(tok0 + r) * DD + lane * 16;
    v8f f0 = *(const v8f*)p;
    v8f f1 = *(const v8f*)(p + 8);
    v8h h0, h1;
#pragma unroll
    for (int i = 0; i < 8; ++i) { h0[i] = (_Float16)f0[i]; h1[i] = (_Float16)f1[i]; }
    *(v8h*)(xw + r * DD + lane * 16)     = h0;
    *(v8h*)(xw + r * DD + lane * 16 + 8) = h1;
  }
  asm volatile("s_wait_dscnt 0" ::: "memory");  // intra-wave store->load

  // =========== fused gating: logits[16 tok][16 cols=(t*8+e)] in f32 =========
  // A 16x4 f32: lane(&15)=row, {K 4kk+2*half, +1}. B 4x16 f32: lane=col, same K.
  v8f ga[4];
#pragma unroll
  for (int q = 0; q < 4; ++q) ga[q] = zero8;
  {
    const float* xr2 = x + (size_t)(tok0 + ln) * DD + 2 * half;  // cache-hot
    const float* wc  = wgt + (size_t)ln * DD + 2 * half;
#pragma unroll 1
    for (int kk2 = 0; kk2 < 32; ++kk2) {
#pragma unroll
      for (int q = 0; q < 4; ++q) {
        int kk = kk2 * 4 + q;
        v2f av = *(const v2f*)(xr2 + kk * 4);
        v2f bv = *(const v2f*)(wc + kk * 4);
        ga[q] = __builtin_amdgcn_wmma_f32_16x16x4_f32(
            false, av, false, bv, (short)0, ga[q], false, false);
      }
    }
  }
  v8f lgv = (ga[0] + ga[1]) + (ga[2] + ga[3]);
  {
    float gbias = bg[ln];                 // ln == t*8+e column index
#pragma unroll
    for (int i = 0; i < 8; ++i) lgv[i] += gbias;
  }

  // --- per-row top-2 via butterfly merge over each 8-lane expert group ---
  const int myE = ln & 7;
  const int tGrp = ln >> 3;               // task of this lane's column
  float g0r[8]; int pidx[8];
#pragma unroll
  for (int i = 0; i < 8; ++i) {
    float v0 = lgv[i]; int i0 = myE;
    float v1 = -3.0e38f; int i1 = 0;
#pragma unroll
    for (int m = 1; m <= 4; m <<= 1) {
      float ov0 = __shfl_xor(v0, m, 32);
      int   oi0 = __shfl_xor(i0, m, 32);
      float ov1 = __shfl_xor(v1, m, 32);
      int   oi1 = __shfl_xor(i1, m, 32);
      if (ov0 > v0) {
        float s = v0; int si = i0;
        v0 = ov0; i0 = oi0;
        if (ov1 > s) { s = ov1; si = oi1; }
        v1 = s; i1 = si;
      } else if (ov0 > v1) { v1 = ov0; i1 = oi0; }
    }
    float gg0 = 1.f / (1.f + __expf(v1 - v0));   // softmax over {v0,v1}
    g0r[i] = gg0;
    pidx[i] = i0 | (i1 << 8);
    if (myE == 0) {   // one acting lane per (task, row-half) group
      atomicAdd(&red[tGrp * 8 + i0], gg0);
      atomicAdd(&red[tGrp * 8 + i1], 1.f - gg0);
      atomicAdd(&red[16 + tGrp * 8 + i0], 1.f);
      atomicAdd(&red[16 + tGrp * 8 + i1], 1.f);
    }
  }
  // broadcast gate descriptors to every lane (rows matching reader's half)
  float g0b[TT][8]; int ib[TT][8];
#pragma unroll
  for (int t = 0; t < TT; ++t) {
    int src = t * 8 + half * 16;
#pragma unroll
    for (int i = 0; i < 8; ++i) {
      g0b[t][i] = __shfl(g0r[i], src, 32);
      ib[t][i]  = __shfl(pidx[i], src, 32);
    }
  }
  __syncthreads();
  if (threadIdx.x < 32)
    partials[(size_t)blockIdx.x * 32 + threadIdx.x] = red[threadIdx.x];

  // ======================= expert MLPs + combine ============================
  float yacc[TT][4][8];
#pragma unroll
  for (int t = 0; t < TT; ++t)
#pragma unroll
    for (int nt = 0; nt < 4; ++nt)
#pragma unroll
      for (int i = 0; i < 8; ++i) yacc[t][nt][i] = 0.f;

#pragma unroll 1
  for (int e = 0; e < EE; ++e) {
    // ---------------- layer 1: [16x512] @ [512x64], pipelined -------------
    v8f acc1[4];
#pragma unroll
    for (int nt = 0; nt < 4; ++nt) acc1[nt] = zero8;
    const _Float16* w1 = W1t + (size_t)e * H1V * DD;

    v16h a_c = loadA16(xw, ln, half, DD, 0);
    v16h bc[4];
#pragma unroll
    for (int nt = 0; nt < 4; ++nt) bc[nt] = loadB16(w1, DD, nt, ln, half, 0);

#pragma unroll
    for (int c = 0; c < 16; ++c) {
      v16h a_n; v16h bn[4];
      if (c < 15) {                    // prefetch next chunk (A from LDS,
        a_n = loadA16(xw, ln, half, DD, c + 1);   //  4x B from global)
#pragma unroll
        for (int nt = 0; nt < 4; ++nt) bn[nt] = loadB16(w1, DD, nt, ln, half, c + 1);
      } else {
        a_n = a_c;
#pragma unroll
        for (int nt = 0; nt < 4; ++nt) bn[nt] = bc[nt];
      }
      __builtin_amdgcn_sched_barrier(0);  // keep prefetch above the WMMAs
#pragma unroll
      for (int nt = 0; nt < 4; ++nt)
        acc1[nt] = __builtin_amdgcn_wmma_f32_16x16x32_f16(
            false, a_c, false, bc[nt], (short)0, acc1[nt], false, false);
      __builtin_amdgcn_sched_barrier(0);
      a_c = a_n;
#pragma unroll
      for (int nt = 0; nt < 4; ++nt) bc[nt] = bn[nt];
    }
    // bias + relu, C-layout (lane=N, vgpr=M) -> row-major f16 tile in LDS
#pragma unroll
    for (int nt = 0; nt < 4; ++nt) {
      float bias = b1[e * H1V + nt * 16 + ln];
#pragma unroll
      for (int i = 0; i < 8; ++i) {
        float v = acc1[nt][i] + bias;
        v = v > 0.f ? v : 0.f;
        h1s[(i + half * 8) * H1V + nt * 16 + ln] = (_Float16)v;
      }
    }
    asm volatile("s_wait_dscnt 0" ::: "memory");

    // ---------------- layer 2: [16x64] @ [64x32] ----------------
    v8f acc2[2];
    acc2[0] = zero8; acc2[1] = zero8;
    const _Float16* w2 = W2t + (size_t)e * H2V * H1V;
#pragma unroll
    for (int c = 0; c < 2; ++c) {
      v16h a = loadA16(h1s, ln, half, H1V, c);
#pragma unroll
      for (int nt = 0; nt < 2; ++nt) {
        v16h bf = loadB16(w2, H1V, nt, ln, half, c);
        acc2[nt] = __builtin_amdgcn_wmma_f32_16x16x32_f16(
            false, a, false, bf, (short)0, acc2[nt], false, false);
      }
    }
#pragma unroll
    for (int nt = 0; nt < 2; ++nt) {
      float bias = b2[e * H2V + nt * 16 + ln];
#pragma unroll
      for (int i = 0; i < 8; ++i) {
        float v = acc2[nt][i] + bias;
        v = v > 0.f ? v : 0.f;
        h2s[(i + half * 8) * H2V + nt * 16 + ln] = (_Float16)v;
      }
    }
    asm volatile("s_wait_dscnt 0" ::: "memory");

    // ---------------- layer 3: [16x32] @ [32x64] ----------------
    v16h a3 = loadA16(h2s, ln, half, H2V, 0);
    const _Float16* w3 = W3t + (size_t)e * OO * H2V;
    v8f acc3[4];
#pragma unroll
    for (int nt = 0; nt < 4; ++nt) acc3[nt] = zero8;
#pragma unroll
    for (int nt = 0; nt < 4; ++nt) {
      v16h bf = loadB16(w3, H2V, nt, ln, half, 0);
      acc3[nt] = __builtin_amdgcn_wmma_f32_16x16x32_f16(
          false, a3, false, bf, (short)0, acc3[nt], false, false);
    }

    // gate values from in-register descriptors (no memory traffic)
    float gv[TT][8];
#pragma unroll
    for (int t = 0; t < TT; ++t)
#pragma unroll
      for (int i = 0; i < 8; ++i) {
        int p = ib[t][i];
        float gg = g0b[t][i];
        gv[t][i] = (e == (p & 0xff)) ? gg : ((e == (p >> 8)) ? 1.f - gg : 0.f);
      }

#pragma unroll
    for (int nt = 0; nt < 4; ++nt) {
      float bias = b3[e * OO + nt * 16 + ln];
#pragma unroll
      for (int i = 0; i < 8; ++i) {
        float v = acc3[nt][i] + bias;
        v = v > 0.f ? v : 0.f;
#pragma unroll
        for (int t = 0; t < TT; ++t) yacc[t][nt][i] += gv[t][i] * v;
      }
    }
  }

  // ---- write y [T,B,O]: lane = col, vgpr = row (C layout) ----
#pragma unroll
  for (int t = 0; t < TT; ++t)
#pragma unroll
    for (int nt = 0; nt < 4; ++nt)
#pragma unroll
      for (int i = 0; i < 8; ++i)
        y[((size_t)t * BB + tok0 + i + half * 8) * OO + nt * 16 + ln] = yacc[t][nt][i];
}

// ---------------------------------------------------------------------------
// Kernel 3: reduce per-block partials and compute cv^2 loss -> d_out[T*B*O]
// ---------------------------------------------------------------------------
__global__ void reduce_loss_kernel(const float* __restrict__ partials,
                                   float* __restrict__ out)
{
  __shared__ float sums[32];
  int j = threadIdx.x;
  if (j < 32) {
    float s = 0.f;
    for (int b = 0; b < NBLK; ++b) s += partials[(size_t)b * 32 + j];
    sums[j] = s;
  }
  __syncthreads();
  if (j == 0) {
    float total = 0.f;
    for (int which = 0; which < 2; ++which) {
      for (int t = 0; t < TT; ++t) {
        const float* v = sums + which * 16 + t * 8;
        float mean = 0.f;
        for (int e = 0; e < EE; ++e) mean += v[e];
        mean *= (1.f / EE);
        float var = 0.f;
        for (int e = 0; e < EE; ++e) { float d = v[e] - mean; var += d * d; }
        var *= (1.f / (EE - 1));   // unbiased (ddof=1)
        total += var / (mean * mean + 1e-10f);
      }
    }
    out[0] = 0.01f * total;
  }
}

// ---------------------------------------------------------------------------
extern "C" void kernel_launch(void* const* d_in, const int* in_sizes, int n_in,
                              void* d_out, int out_size, void* d_ws, size_t ws_size,
                              hipStream_t stream) {
  const float* x  = (const float*)d_in[0];
  const float* wg = (const float*)d_in[1];
  const float* bg = (const float*)d_in[2];
  const float* W1 = (const float*)d_in[3];
  const float* b1 = (const float*)d_in[4];
  const float* W2 = (const float*)d_in[5];
  const float* b2 = (const float*)d_in[6];
  const float* W3 = (const float*)d_in[7];
  const float* b3 = (const float*)d_in[8];
  float* y = (float*)d_out;

  // Workspace layout (~0.9 MB)
  char* ws = (char*)d_ws;
  _Float16* W1t = (_Float16*)ws;                       // 512 KB
  _Float16* W2t = W1t + (size_t)EE * H1V * DD;         //  32 KB
  _Float16* W3t = W2t + (size_t)EE * H2V * H1V;        //  32 KB
  float*    wgt = (float*)(W3t + (size_t)EE * OO * H2V);          // 32 KB f32
  float*    partials = wgt + (size_t)TT * EE * DD;     // NBLK*32 f32 = 256 KB

  convert_weights<<<(EE * DD * H1V + 255) / 256, 256, 0, stream>>>(
      W1, W2, W3, wg, W1t, W2t, W3t, wgt);

  moe_fused_kernel<<<NBLK, WPB * 32, 0, stream>>>(
      x, wgt, bg, W1t, b1, W2t, b2, W3t, b3, y, partials);

  reduce_loss_kernel<<<1, 32, 0, stream>>>(partials, y + (size_t)TT * BB * OO);
}
